// CustomLSTM_72335839199508
// MI455X (gfx1250) — matrix operand order repaired and checked by reading, not compiled, not run on hardware
//
#include <hip/hip_runtime.h>

// Problem constants (match reference)
#define SEQ   512
#define BATCH 64
#define INF_  1024           // input features (== HS, both 1024)
#define HS    1024           // hidden size
#define G4    (4 * HS)       // 4096 gate columns
#define BH    (BATCH * HS)   // 65536 elements per [B,H] slab
#define NBLK  32             // persistent workgroups per layer kernel
#define SK    64             // staged K-chunk width
#define NSTG  (INF_ / SK)    // 16 stages per phase

// Async global->LDS DMA path (gfx1250): guarded so absence falls back cleanly.
#if defined(__has_builtin)
# if __has_builtin(__builtin_amdgcn_global_load_async_to_lds_b128)
#  define USE_ASYNC_LDS 1
# endif
#endif
#ifndef USE_ASYNC_LDS
# define USE_ASYNC_LDS 0
#endif

typedef __attribute__((ext_vector_type(16))) __bf16          bf16x16;
typedef __attribute__((ext_vector_type(8)))  float           v8f;
typedef __attribute__((ext_vector_type(8)))  unsigned short  u16x8;
typedef __attribute__((ext_vector_type(16))) unsigned short  u16x16;

__device__ __forceinline__ unsigned short f2bf(float f) {
  union { float f; unsigned u; } v; v.f = f;
  unsigned r = v.u + 0x7FFFu + ((v.u >> 16) & 1u);  // round-to-nearest-even
  return (unsigned short)(r >> 16);
}

__device__ __forceinline__ float sigmoidf_(float x) {
  return 1.0f / (1.0f + __expf(-x));
}

#if USE_ASYNC_LDS
// builtin parameter type (from hipcc diagnostic): 16-byte int vector, AS1 / AS3
typedef int vsi4 __attribute__((vector_size(16)));
typedef __attribute__((address_space(1))) vsi4* g_v4p;
typedef __attribute__((address_space(3))) vsi4* l_v4p;

// one b128 (16B = 8 bf16) DMA from global to LDS, ASYNCcnt-tracked
__device__ __forceinline__ void async_g2l(const unsigned short* g, unsigned short* l) {
  __builtin_amdgcn_global_load_async_to_lds_b128(
      (g_v4p)(void*)g, (l_v4p)(void*)l, 0, 0);
}
__device__ __forceinline__ void wait_async0() {
#if defined(__has_builtin) && __has_builtin(__builtin_amdgcn_s_wait_asynccnt)
  __builtin_amdgcn_s_wait_asynccnt(0);
#else
  asm volatile("s_wait_asynccnt 0x0" ::: "memory");
#endif
}
#endif

// ---------------- fp32 -> bf16 bulk conversion ----------------
__global__ void cvt_f32_bf16(const float* __restrict__ src,
                             unsigned short* __restrict__ dst, int n) {
  int i = blockIdx.x * blockDim.x + threadIdx.x;
  int stride = gridDim.x * blockDim.x;
  for (; i < n; i += stride) dst[i] = f2bf(src[i]);
}

// ---------------- zero-init of recurrent state + barrier counters ----------------
__global__ void init_state(float* __restrict__ c0, float* __restrict__ c1,
                           unsigned short* __restrict__ hA0,
                           unsigned short* __restrict__ hA1,
                           unsigned int* __restrict__ cnt, int ncnt) {
  int i = blockIdx.x * blockDim.x + threadIdx.x;   // grid covers BH exactly
  c0[i] = 0.0f; c1[i] = 0.0f; hA0[i] = 0; hA1[i] = 0;
  if (i < ncnt) cnt[i] = 0;
}

// ---------------- persistent fused LSTM layer ----------------
// One launch runs all 512 timesteps. 32 blocks x 256 threads; block b owns
// columns [32b, 32b+32) of H and its cell-state slice (never leaves the block).
// Per step: gates = x_t @ U^T + h_{t-1} @ V^T (K = 1024+1024), double-buffered
// K=64 LDS stages (async DMA when available), 8 waves = 4 m-tiles x 2 n-tiles,
// 4 WMMA accumulators (i,f,g,o). h broadcast in bf16 via L2; one fenced atomic
// grid-barrier per step.
__launch_bounds__(256)
__global__ void lstm_layer(const unsigned short* __restrict__ xseq, // [SEQ,BATCH,1024] bf16
                           const unsigned short* __restrict__ Ubf,  // [4H, 1024] bf16
                           const unsigned short* __restrict__ Vbf,  // [4H, 1024] bf16
                           const float* __restrict__ bih,
                           const float* __restrict__ bhh,
                           float* __restrict__ c,                   // [BATCH, HS] slice-owned
                           unsigned short* __restrict__ hA,         // ping (zeroed)
                           unsigned short* __restrict__ hB,         // pong
                           unsigned short* __restrict__ hseq_bf,    // [SEQ,BATCH,HS] or null
                           float* __restrict__ hf_base,             // fp32 h out
                           int hf_stride,                            // 0 or BH
                           unsigned int* __restrict__ bar)           // [SEQ] counters (zeroed)
{
  __shared__ unsigned short ldsA[2][64 * SK];    // 16 KB: A stage (64 rows x 64 k)
  __shared__ unsigned short ldsB[2][128 * SK];   // 32 KB: B stage (4 gates x 32 cols x 64 k)

  const int tid    = threadIdx.x;
  const int lane   = tid & 31;
  const int laneLo = lane & 15;
  const int laneHi = lane >> 4;
  const int wave   = tid >> 5;
  const int mbase  = (wave >> 1) * 16;           // 0,16,32,48
  const int ntL    = wave & 1;                   // n-tile within block
  const int nblock = blockIdx.x * 32;
  const int kOffA  = laneHi * 16;                // contiguous k = 16h + s (matches B)
  const int kOffB  = laneHi * 16;                // contiguous k = 16h + s (matches A)

  // cooperative staging coordinates (256 threads)
  const int sArow = tid >> 2, sAk = (tid & 3) * 16;       // 16 elems (32B) each
  const int sBrow = tid >> 1, sBk = (tid & 1) * 32;       // 32 elems (64B) each
  const size_t sBgRow = (size_t)((sBrow >> 5) * HS + nblock + (sBrow & 31));
  unsigned short* lA0 = &ldsA[0][sArow * SK + sAk];
  unsigned short* lA1 = &ldsA[1][sArow * SK + sAk];
  unsigned short* lB0 = &ldsB[0][sBrow * SK + sBk];
  unsigned short* lB1 = &ldsB[1][sBrow * SK + sBk];

  // per-lane biases (hoisted out of the time loop)
  const int ncol = nblock + ntL * 16 + laneLo;
  const float bi = bih[ncol]          + bhh[ncol];
  const float bf = bih[HS + ncol]     + bhh[HS + ncol];
  const float bg = bih[2 * HS + ncol] + bhh[2 * HS + ncol];
  const float bo = bih[3 * HS + ncol] + bhh[3 * HS + ncol];

#pragma unroll 1
  for (int t = 0; t < SEQ; ++t) {
    const unsigned short* hprev = (t & 1) ? hB : hA;
    unsigned short*       hcur  = (t & 1) ? hA : hB;
    const unsigned short* xt    = xseq + (size_t)t * BH;

    v8f acc0 = {}, acc1 = {}, acc2 = {}, acc3 = {};

#pragma unroll 1
    for (int phase = 0; phase < 2; ++phase) {
      const unsigned short* Asrc = phase ? hprev : xt;
      const unsigned short* Bsrc = phase ? Vbf : Ubf;
      const unsigned short* aG = Asrc + (size_t)sArow * INF_ + sAk;
      const unsigned short* bG = Bsrc + sBgRow * INF_ + sBk;

      // ---- prologue: stage 0 into buf0 ----
#if USE_ASYNC_LDS
      __syncthreads();                         // prior readers of buf0 done
      async_g2l(aG,      lA0);
      async_g2l(aG + 8,  lA0 + 8);
      async_g2l(bG,      lB0);
      async_g2l(bG + 8,  lB0 + 8);
      async_g2l(bG + 16, lB0 + 16);
      async_g2l(bG + 24, lB0 + 24);
      wait_async0();
      __syncthreads();
#else
      u16x16 aReg  = *(const u16x16*)(aG);
      u16x16 bReg0 = *(const u16x16*)(bG);
      u16x16 bReg1 = *(const u16x16*)(bG + 16);
      __syncthreads();                         // prior readers of buf0 done
      *(u16x16*)lA0        = aReg;
      *(u16x16*)lB0        = bReg0;
      *(u16x16*)(lB0 + 16) = bReg1;
      __syncthreads();
#endif

#pragma unroll 2
      for (int s = 0; s < NSTG; ++s) {
        const int cur = s & 1;
        unsigned short* lAn = (cur ? lA0 : lA1);
        unsigned short* lBn = (cur ? lB0 : lB1);
#if USE_ASYNC_LDS
        if (s < NSTG - 1) {                    // DMA next stage straight into LDS
          __syncthreads();                     // everyone done reading buf[nxt]
          const unsigned short* ga = aG + (s + 1) * SK;
          const unsigned short* gb = bG + (s + 1) * SK;
          async_g2l(ga,      lAn);
          async_g2l(ga + 8,  lAn + 8);
          async_g2l(gb,      lBn);
          async_g2l(gb + 8,  lBn + 8);
          async_g2l(gb + 16, lBn + 16);
          async_g2l(gb + 24, lBn + 24);
        }
#else
        u16x16 aN = {}, bN0 = {}, bN1 = {};
        if (s < NSTG - 1) {                    // prefetch next stage into regs
          aN  = *(const u16x16*)(aG + (s + 1) * SK);
          bN0 = *(const u16x16*)(bG + (s + 1) * SK);
          bN1 = *(const u16x16*)(bG + (s + 1) * SK + 16);
        }
#endif
        // ---- compute: two K=32 sub-chunks, 8 WMMAs, all from LDS ----
#pragma unroll
        for (int half = 0; half < 2; ++half) {
          const int ksub = half * 32;
          const unsigned short* la = &ldsA[cur][(mbase + laneLo) * SK + ksub + kOffA];
          u16x8 a0 = *(const u16x8*)(la);
          u16x8 a1 = *(const u16x8*)(la + 8);
          u16x16 afu;
#pragma unroll
          for (int i = 0; i < 8; ++i) { afu[i] = a0[i]; afu[i + 8] = a1[i]; }
          bf16x16 A = __builtin_bit_cast(bf16x16, afu);
          // gate stride in LDS = 32 rows * SK elems = 2048
          const unsigned short* lb = &ldsB[cur][(ntL * 16 + laneLo) * SK + ksub + kOffB];
          bf16x16 B0 = __builtin_bit_cast(bf16x16, *(const u16x16*)(lb));
          bf16x16 B1 = __builtin_bit_cast(bf16x16, *(const u16x16*)(lb + 2048));
          bf16x16 B2 = __builtin_bit_cast(bf16x16, *(const u16x16*)(lb + 4096));
          bf16x16 B3 = __builtin_bit_cast(bf16x16, *(const u16x16*)(lb + 6144));

          acc0 = __builtin_amdgcn_wmma_f32_16x16x32_bf16(false, A, false, B0, (short)0, acc0, false, false);
          acc1 = __builtin_amdgcn_wmma_f32_16x16x32_bf16(false, A, false, B1, (short)0, acc1, false, false);
          acc2 = __builtin_amdgcn_wmma_f32_16x16x32_bf16(false, A, false, B2, (short)0, acc2, false, false);
          acc3 = __builtin_amdgcn_wmma_f32_16x16x32_bf16(false, A, false, B3, (short)0, acc3, false, false);
        }
        if (s < NSTG - 1) {
#if USE_ASYNC_LDS
          wait_async0();                       // DMA landed
          __syncthreads();
#else
          __syncthreads();                     // everyone done reading buf[nxt]
          *(u16x16*)lAn        = aN;
          *(u16x16*)lBn        = bN0;
          *(u16x16*)(lBn + 16) = bN1;
          __syncthreads();
#endif
        }
      }
    }

    // ---- epilogue: gate nonlinearities + cell update (block-owned c slice) ----
    float* hf = hf_base + (size_t)t * hf_stride;
    unsigned short* hs = hseq_bf ? (hseq_bf + (size_t)t * BH) : (unsigned short*)0;
#pragma unroll
    for (int r = 0; r < 8; ++r) {
      const int mrow = mbase + laneHi * 8 + r;           // C/D layout: VGPR r -> M
      const size_t idx = (size_t)mrow * HS + ncol;
      const float iv = sigmoidf_(acc0[r] + bi);
      const float fv = sigmoidf_(acc1[r] + bf);
      const float gv = tanhf(acc2[r] + bg);
      const float ov = sigmoidf_(acc3[r] + bo);
      const float cn = fv * c[idx] + iv * gv;
      const float hv = ov * tanhf(cn);
      const unsigned short hb = f2bf(hv);
      c[idx]    = cn;
      hf[idx]   = hv;
      hcur[idx] = hb;
      if (hs) hs[idx] = hb;
    }

    // ---- grid-wide barrier: all h(t) writes visible before anyone starts t+1 ----
    __syncthreads();
    if (tid == 0) {
      __threadfence();
      atomicAdd(&bar[t], 1u);
      volatile unsigned int* vc = &bar[t];
      while (*vc < (unsigned)gridDim.x) { __builtin_amdgcn_s_sleep(2); }
      __threadfence();
    }
    __syncthreads();
  }
}

extern "C" void kernel_launch(void* const* d_in, const int* in_sizes, int n_in,
                              void* d_out, int out_size, void* d_ws, size_t ws_size,
                              hipStream_t stream) {
  (void)in_sizes; (void)n_in; (void)out_size; (void)ws_size;
  const float* x    = (const float*)d_in[0];
  const float* U0   = (const float*)d_in[1];
  const float* V0   = (const float*)d_in[2];
  const float* bih0 = (const float*)d_in[3];
  const float* bhh0 = (const float*)d_in[4];
  const float* U1   = (const float*)d_in[5];
  const float* V1   = (const float*)d_in[6];
  const float* bih1 = (const float*)d_in[7];
  const float* bhh1 = (const float*)d_in[8];
  float* out = (float*)d_out;

  // Workspace layout: bf16 slabs, then fp32 slabs, then barrier counters (~169 MB)
  unsigned short* ws16 = (unsigned short*)d_ws;
  size_t off = 0;
  const size_t WSZ = (size_t)G4 * INF_;                   // 4M elems per weight
  unsigned short* Ubf0 = ws16 + off; off += WSZ;
  unsigned short* Vbf0 = ws16 + off; off += WSZ;
  unsigned short* Ubf1 = ws16 + off; off += WSZ;
  unsigned short* Vbf1 = ws16 + off; off += WSZ;
  unsigned short* Xbf  = ws16 + off; off += (size_t)SEQ * BH;  // input, bf16
  unsigned short* H0bf = ws16 + off; off += (size_t)SEQ * BH;  // layer0 h seq, bf16
  unsigned short* hA0  = ws16 + off; off += BH;                // layer0 h ping (zeroed)
  unsigned short* hB0  = ws16 + off; off += BH;                // layer0 h pong
  unsigned short* hA1  = ws16 + off; off += BH;                // layer1 h ping (zeroed)
  unsigned short* hB1  = ws16 + off; off += BH;                // layer1 h pong
  float* wsf = (float*)(ws16 + off);
  float* c0  = wsf;              // layer0 cell state
  float* c1  = wsf + BH;         // layer1 cell state
  float* h0f = wsf + 2 * BH;     // layer0 h fp32 (final value survives)
  unsigned int* cnt0 = (unsigned int*)(wsf + 3 * BH);  // [SEQ] barrier counters, layer0
  unsigned int* cnt1 = cnt0 + SEQ;                     // [SEQ] barrier counters, layer1

  // 1) convert inputs/weights to bf16; zero state + barrier counters
  cvt_f32_bf16<<<4096, 256, 0, stream>>>(x,  Xbf,  SEQ * BATCH * INF_);
  cvt_f32_bf16<<<2048, 256, 0, stream>>>(U0, Ubf0, G4 * INF_);
  cvt_f32_bf16<<<2048, 256, 0, stream>>>(V0, Vbf0, G4 * HS);
  cvt_f32_bf16<<<2048, 256, 0, stream>>>(U1, Ubf1, G4 * INF_);
  cvt_f32_bf16<<<2048, 256, 0, stream>>>(V1, Vbf1, G4 * HS);
  init_state<<<BH / 256, 256, 0, stream>>>(c0, c1, hA0, hA1, cnt0, 2 * SEQ);

  // 2) layer 0 (persistent, all 512 steps): h seq kept in bf16 for layer 1
  lstm_layer<<<NBLK, 256, 0, stream>>>(Xbf, Ubf0, Vbf0, bih0, bhh0,
                                       c0, hA0, hB0, H0bf, h0f, 0, cnt0);

  // 3) layer 1 (persistent): fp32 h written straight into out1 region
  lstm_layer<<<NBLK, 256, 0, stream>>>(H0bf, Ubf1, Vbf1, bih1, bhh1,
                                       c1, hA1, hB1, (unsigned short*)0,
                                       out, BH, cnt1);

  // 4) assemble h_f [2,B,H] and c_f [2,B,H] after out1
  const size_t OUT1 = (size_t)SEQ * BH;
  (void)hipMemcpyAsync(out + OUT1,          h0f,                          BH * sizeof(float),
                       hipMemcpyDeviceToDevice, stream);
  (void)hipMemcpyAsync(out + OUT1 + BH,     out + (size_t)(SEQ - 1) * BH, BH * sizeof(float),
                       hipMemcpyDeviceToDevice, stream);
  (void)hipMemcpyAsync(out + OUT1 + 2 * BH, c0,                           BH * sizeof(float),
                       hipMemcpyDeviceToDevice, stream);
  (void)hipMemcpyAsync(out + OUT1 + 3 * BH, c1,                           BH * sizeof(float),
                       hipMemcpyDeviceToDevice, stream);
}
